// SparseTransformer_Model_38594576122324
// MI455X (gfx1250) — compile-verified, matrix operand order checked by reading, not run on hardware
//
#include <hip/hip_runtime.h>
#include <hip/hip_bf16.h>
#include <stdint.h>

// ---------------------------------------------------------------------------
// CDNA5 (gfx1250, wave32) WMMA types
// ---------------------------------------------------------------------------
typedef __bf16 bf16_t;
typedef __attribute__((ext_vector_type(16))) __bf16 v16bf;
typedef __attribute__((ext_vector_type(8)))  __bf16 v8bf;
typedef __attribute__((ext_vector_type(4)))  __bf16 v4bf;
typedef __attribute__((ext_vector_type(8)))  float  v8f;

#define D_MODEL   512
#define NUM_HEADS 8
#define DFF       2048
#define BATCH     32
#define HIST_LEN  512
#define F_LEN     128

#define USE_ASYNC 1   // gfx1250 async global->LDS copies for straight-copy tiles

// ---------------------------------------------------------------------------
// Helpers
// ---------------------------------------------------------------------------
__device__ __forceinline__ bf16_t f2bf(float f) {
  union { float f; uint32_t u; } x; x.f = f;
  uint32_t r = (x.u + 0x7FFFu + ((x.u >> 16) & 1u)) >> 16; // RNE
  union { uint16_t s; bf16_t b; } y; y.s = (uint16_t)r;
  return y.b;
}
__device__ __forceinline__ float bf2f(bf16_t b) {
  union { bf16_t b; uint16_t s; } x; x.b = b;
  union { uint32_t u; float f; } y; y.u = ((uint32_t)x.s) << 16;
  return y.f;
}

// 16-element bf16 fragment: 8 contiguous at p, 8 contiguous at p+16 elements.
// Matches the 16-bit A-matrix 16x32 per-lane layout (CDNA5 ISA);
// B is staged transposed in LDS so it uses the identical pattern.
__device__ __forceinline__ v16bf ld_frag(const bf16_t* p) {
  v16bf r;
#pragma unroll
  for (int i = 0; i < 8; ++i) { r[i] = p[i]; r[i + 8] = p[i + 16]; }
  return r;
}

__device__ __forceinline__ v8f zero8() {
  v8f z;
#pragma unroll
  for (int i = 0; i < 8; ++i) z[i] = 0.f;
  return z;
}

__device__ __forceinline__ float selu_f(float v) {
  const float l = 1.0507009873554805f, a = 1.6732632423543772f;
  return v > 0.f ? l * v : l * a * (__expf(v) - 1.f);
}

// gfx1250 async global->LDS 16-byte copy (VGLOBAL op 98, tracked by ASYNCcnt).
__device__ __forceinline__ void async_cp16(const bf16_t* gsrc, bf16_t* ldst) {
#if USE_ASYNC
  uint32_t l = (uint32_t)(size_t)ldst;   // flat LDS addr[31:0] == LDS offset
  asm volatile("global_load_async_to_lds_b128 %0, %1, off"
               :: "v"(l), "v"(gsrc) : "memory");
#else
  *(uint4*)ldst = *(const uint4*)gsrc;
#endif
}
__device__ __forceinline__ void async_wait() {
#if USE_ASYNC
  asm volatile("s_wait_asynccnt 0x0" ::: "memory");
#endif
}

// ---------------------------------------------------------------------------
// bf16 GEMM: C[M,N] = act( shiftA[M,K] @ B[K,N] + bias (+ Cin) )
// A rows logically shifted within each pow2 sequence (causal conv as 3 GEMMs).
// 64x64 tile / 128 threads (4 waves); K staged 64 deep; bf16 in/out, f32 acc.
// All real dims are multiples of 64, so tile copies are unconditional.
// ---------------------------------------------------------------------------
__global__ void __launch_bounds__(128)
gemm_bf16(const bf16_t* __restrict__ A, int lda, int seqMask, int shift,
          const bf16_t* __restrict__ Bw, int ldb,
          const float* __restrict__ bias,
          const bf16_t* __restrict__ Cin,   // accumulate source (may alias Cout)
          bf16_t* __restrict__ Cout, int ldc,
          int M, int N, int K, int act)
{
  __shared__ bf16_t Alds[64 * 64];
  __shared__ bf16_t Blds[64 * 64];   // Blds[n][k]

  const int tid  = threadIdx.x;
  const int lane = tid & 31;
  const int wave = tid >> 5;
  const int m0   = blockIdx.y * 64;
  const int n0   = blockIdx.x * 64;
  const int h    = lane >> 4;
  const int mm   = lane & 15;

  v8f acc[4];
#pragma unroll
  for (int t = 0; t < 4; ++t) acc[t] = zero8();

  for (int k0 = 0; k0 < K; k0 += 64) {
    // ---- stage A tile (64 rows x 64 k), 16B chunks, 4 per thread ----
    if (shift == 0) {
#pragma unroll
      for (int it = 0; it < 4; ++it) {
        int idx = it * 128 + tid;
        int m = idx >> 3, c = (idx & 7) * 8;
        async_cp16(A + (size_t)(m0 + m) * lda + k0 + c, &Alds[m * 64 + c]);
      }
    } else {
#pragma unroll
      for (int it = 0; it < 4; ++it) {
        int idx = it * 128 + tid;
        int m = idx >> 3, c = (idx & 7) * 8;
        int gr = m0 + m;
        int s = gr & seqMask;
        uint4 val = {0u, 0u, 0u, 0u};
        if (s + shift >= 0)
          val = *(const uint4*)(A + (size_t)(gr + shift) * lda + k0 + c);
        *(uint4*)&Alds[m * 64 + c] = val;
      }
    }
    // ---- stage B tile transposed: Blds[n][k] = B[k0+k][n0+n] ----
#pragma unroll
    for (int it = 0; it < 4; ++it) {
      int idx = it * 128 + tid;
      int k = idx >> 3, n8 = (idx & 7) * 8;
      v8bf bv = *(const v8bf*)(Bw + (size_t)(k0 + k) * ldb + n0 + n8);
#pragma unroll
      for (int j = 0; j < 8; ++j) Blds[(n8 + j) * 64 + k] = bv[j];
    }
    async_wait();
    __syncthreads();

    // ---- 8 WMMAs per stage: each wave computes a 16x64 strip ----
#pragma unroll
    for (int ks = 0; ks < 64; ks += 32) {
      v16bf a = ld_frag(&Alds[(wave * 16 + mm) * 64 + ks + h * 8]);
#pragma unroll
      for (int nt = 0; nt < 4; ++nt) {
        v16bf bb = ld_frag(&Blds[(nt * 16 + mm) * 64 + ks + h * 8]);
        acc[nt] = __builtin_amdgcn_wmma_f32_16x16x32_bf16(
            false, a, false, bb, (short)0, acc[nt], false, false);
      }
    }
    __syncthreads();
  }

  // ---- epilogue: bias / accumulate / activation, bf16 out ----
#pragma unroll
  for (int nt = 0; nt < 4; ++nt) {
    int n = n0 + nt * 16 + mm;
    float bv = bias ? bias[n] : 0.f;
#pragma unroll
    for (int v = 0; v < 8; ++v) {
      int m = m0 + wave * 16 + v + 8 * h;
      float val = acc[nt][v] + bv;
      if (Cin) val += bf2f(Cin[(size_t)m * ldc + n]);
      if (act) val = selu_f(val);
      Cout[(size_t)m * ldc + n] = f2bf(val);
    }
  }
}

// ---------------------------------------------------------------------------
// Dense attention (flash style): one block = 64 queries of one (b,h).
// Q,K,V,O are bf16 [B, S, D_MODEL], head hh at columns hh*64..hh*64+63.
// ---------------------------------------------------------------------------
__global__ void __launch_bounds__(128)
attn_dense(const bf16_t* __restrict__ Q, const bf16_t* __restrict__ Kb,
           const bf16_t* __restrict__ Vb, bf16_t* __restrict__ O,
           int Sq, int Skv, int causal)
{
  __shared__ bf16_t Qs[64 * 64];
  __shared__ bf16_t Ks[64 * 64];     // Ks[j][d]
  __shared__ bf16_t Vs[64 * 64];     // Vs[d][j] (transposed for B fragment)
  __shared__ float  Ss[64 * 64];
  __shared__ bf16_t Ps[64 * 64];
  __shared__ float  rowmax[64], rowsum[64], rescale[64];

  const int tid  = threadIdx.x;
  const int lane = tid & 31;
  const int wave = tid >> 5;
  const int h    = lane >> 4;
  const int mm   = lane & 15;
  const int bh   = blockIdx.x;
  const int b    = bh >> 3;
  const int hh   = bh & 7;
  const int q0   = blockIdx.y * 64;
  const int hOff = hh * 64;
  const float scale = 0.125f;        // 1/sqrt(64)

  // stage Q block (straight copy -> async)
#pragma unroll
  for (int it = 0; it < 4; ++it) {
    int idx = it * 128 + tid;
    int m = idx >> 3, c = (idx & 7) * 8;
    async_cp16(Q + (size_t)(b * Sq + q0 + m) * D_MODEL + hOff + c, &Qs[m * 64 + c]);
  }
  if (tid < 64) { rowmax[tid] = -1e30f; rowsum[tid] = 0.f; }

  v8f acc[4];
#pragma unroll
  for (int t = 0; t < 4; ++t) acc[t] = zero8();
  async_wait();
  __syncthreads();

  const int nkb = (Skv + 63) >> 6;
  for (int kb = 0; kb < nkb; ++kb) {
    if (causal && kb * 64 > q0 + 63) break;

    // stage K (async copy) and V (manual transpose)
#pragma unroll
    for (int it = 0; it < 4; ++it) {
      int idx = it * 128 + tid;
      int j = idx >> 3, c = (idx & 7) * 8;
      async_cp16(Kb + (size_t)(b * Skv + kb * 64 + j) * D_MODEL + hOff + c,
                 &Ks[j * 64 + c]);
      v8bf vv = *(const v8bf*)(Vb + (size_t)(b * Skv + kb * 64 + j) * D_MODEL + hOff + c);
#pragma unroll
      for (int e = 0; e < 8; ++e) Vs[(c + e) * 64 + j] = vv[e];
    }
    async_wait();
    __syncthreads();

    // S = scale * Q K^T (+ mask) -> Ss
#pragma unroll
    for (int nt = 0; nt < 4; ++nt) {
      v8f c = zero8();
#pragma unroll
      for (int ks = 0; ks < 64; ks += 32) {
        v16bf a  = ld_frag(&Qs[(wave * 16 + mm) * 64 + ks + h * 8]);
        v16bf bb = ld_frag(&Ks[(nt * 16 + mm) * 64 + ks + h * 8]);
        c = __builtin_amdgcn_wmma_f32_16x16x32_bf16(
            false, a, false, bb, (short)0, c, false, false);
      }
#pragma unroll
      for (int v = 0; v < 8; ++v) {
        int m = wave * 16 + v + 8 * h;
        int n = nt * 16 + mm;
        int jg = kb * 64 + n;
        float s = c[v] * scale;
        if (causal && jg > q0 + m) s = -1e9f;
        Ss[m * 64 + n] = s;
      }
    }
    __syncthreads();

    // online softmax stats (one thread per query row)
    if (tid < 64) {
      int r = tid;
      float tmax = -1e30f;
      for (int j = 0; j < 64; ++j) tmax = fmaxf(tmax, Ss[r * 64 + j]);
      float om = rowmax[r];
      float nm = fmaxf(om, tmax);
      float rs = __expf(om - nm);
      float psum = 0.f;
      for (int j = 0; j < 64; ++j) {
        float p = __expf(Ss[r * 64 + j] - nm);
        Ps[r * 64 + j] = f2bf(p);
        psum += p;
      }
      rowmax[r]  = nm;
      rowsum[r]  = rowsum[r] * rs + psum;
      rescale[r] = rs;
    }
    __syncthreads();

    // rescale accumulators, then O += P @ V
#pragma unroll
    for (int nt = 0; nt < 4; ++nt) {
#pragma unroll
      for (int v = 0; v < 8; ++v) {
        int m = wave * 16 + v + 8 * h;
        acc[nt][v] = acc[nt][v] * rescale[m];
      }
#pragma unroll
      for (int ks = 0; ks < 64; ks += 32) {
        v16bf a  = ld_frag(&Ps[(wave * 16 + mm) * 64 + ks + h * 8]);
        v16bf bb = ld_frag(&Vs[(nt * 16 + mm) * 64 + ks + h * 8]);
        acc[nt] = __builtin_amdgcn_wmma_f32_16x16x32_bf16(
            false, a, false, bb, (short)0, acc[nt], false, false);
      }
    }
    __syncthreads();
  }

  // write O = acc / rowsum (bf16)
#pragma unroll
  for (int nt = 0; nt < 4; ++nt) {
    int n = nt * 16 + mm;            // depth
#pragma unroll
    for (int v = 0; v < 8; ++v) {
      int m = wave * 16 + v + 8 * h;
      O[(size_t)(b * Sq + q0 + m) * D_MODEL + hOff + n] = f2bf(acc[nt][v] / rowsum[m]);
    }
  }
}

// ---------------------------------------------------------------------------
// Encoder block-sparse attention. Faithful to the reference: off-block logits
// are exactly 0 (NOT -inf). With M = max(rowmax_block, 0):
//   Z = sum_block exp(s-M) + 448*exp(-M)
//   o = ( P @ V_block + exp(-M) * (sumV_all - sumV_block) ) / Z
// blocksum: [B*H, 8, 64] per-block V column sums (f32).
// ---------------------------------------------------------------------------
__global__ void __launch_bounds__(128)
attn_sparse_enc(const bf16_t* __restrict__ Q, const bf16_t* __restrict__ Kb,
                const bf16_t* __restrict__ Vb, const float* __restrict__ blocksum,
                bf16_t* __restrict__ O)
{
  __shared__ bf16_t Qs[64 * 64];
  __shared__ bf16_t Ks[64 * 64];
  __shared__ bf16_t Vs[64 * 64];     // transposed: Vs[d][j]
  __shared__ float  Ss[64 * 64];
  __shared__ bf16_t Ps[64 * 64];
  __shared__ float  invZ[64], ecorr[64], Vcorr[64];

  const int tid  = threadIdx.x;
  const int lane = tid & 31;
  const int wave = tid >> 5;
  const int h    = lane >> 4;
  const int mm   = lane & 15;
  const int bh   = blockIdx.x;
  const int b    = bh >> 3;
  const int hh   = bh & 7;
  const int blk  = blockIdx.y;       // 0..7
  const int r0   = blk * 64;
  const int hOff = hh * 64;
  const float scale = 0.125f;

  // correction vector: sum of all block V-sums minus own block
  if (tid < 64) {
    const float* bs = blocksum + (size_t)bh * 8 * 64;
    float tot = 0.f;
#pragma unroll
    for (int t = 0; t < 8; ++t) tot += bs[t * 64 + tid];
    Vcorr[tid] = tot - bs[blk * 64 + tid];
  }
  // stage Q,K (async) and V (manual transpose) of the diagonal block
#pragma unroll
  for (int it = 0; it < 4; ++it) {
    int idx = it * 128 + tid;
    int m = idx >> 3, c = (idx & 7) * 8;
    size_t row = (size_t)(b * HIST_LEN + r0 + m) * D_MODEL + hOff + c;
    async_cp16(Q + row, &Qs[m * 64 + c]);
    async_cp16(Kb + row, &Ks[m * 64 + c]);
    v8bf vv = *(const v8bf*)(Vb + row);
#pragma unroll
    for (int e = 0; e < 8; ++e) Vs[(c + e) * 64 + m] = vv[e];
  }
  async_wait();
  __syncthreads();

  // S = scale * Q K^T
#pragma unroll
  for (int nt = 0; nt < 4; ++nt) {
    v8f c = zero8();
#pragma unroll
    for (int ks = 0; ks < 64; ks += 32) {
      v16bf a  = ld_frag(&Qs[(wave * 16 + mm) * 64 + ks + h * 8]);
      v16bf bb = ld_frag(&Ks[(nt * 16 + mm) * 64 + ks + h * 8]);
      c = __builtin_amdgcn_wmma_f32_16x16x32_bf16(
          false, a, false, bb, (short)0, c, false, false);
    }
#pragma unroll
    for (int v = 0; v < 8; ++v) {
      int m = wave * 16 + v + 8 * h;
      Ss[m * 64 + nt * 16 + mm] = c[v] * scale;
    }
  }
  __syncthreads();

  // softmax with the 448 zero logits folded in
  if (tid < 64) {
    int r = tid;
    float tmax = -1e30f;
    for (int j = 0; j < 64; ++j) tmax = fmaxf(tmax, Ss[r * 64 + j]);
    float Mx = fmaxf(tmax, 0.f);
    float psum = 0.f;
    for (int j = 0; j < 64; ++j) {
      float p = __expf(Ss[r * 64 + j] - Mx);
      Ps[r * 64 + j] = f2bf(p);
      psum += p;
    }
    float em = __expf(-Mx);
    float Z = psum + 448.f * em;     // 512-64 off-block entries, each exp(0-Mx)
    invZ[r]  = 1.f / Z;
    ecorr[r] = em;
  }
  __syncthreads();

  // O = (P @ V + ecorr * Vcorr) * invZ
  v8f acc[4];
#pragma unroll
  for (int t = 0; t < 4; ++t) acc[t] = zero8();
#pragma unroll
  for (int nt = 0; nt < 4; ++nt) {
#pragma unroll
    for (int ks = 0; ks < 64; ks += 32) {
      v16bf a  = ld_frag(&Ps[(wave * 16 + mm) * 64 + ks + h * 8]);
      v16bf bb = ld_frag(&Vs[(nt * 16 + mm) * 64 + ks + h * 8]);
      acc[nt] = __builtin_amdgcn_wmma_f32_16x16x32_bf16(
          false, a, false, bb, (short)0, acc[nt], false, false);
    }
    int n = nt * 16 + mm;            // depth
#pragma unroll
    for (int v = 0; v < 8; ++v) {
      int m = wave * 16 + v + 8 * h;
      float val = (acc[nt][v] + ecorr[m] * Vcorr[n]) * invZ[m];
      O[(size_t)(b * HIST_LEN + r0 + m) * D_MODEL + hOff + n] = f2bf(val);
    }
  }
}

// ---------------------------------------------------------------------------
// Per-block V column sums for the encoder correction. grid=B*H, block=64.
// ---------------------------------------------------------------------------
__global__ void __launch_bounds__(64)
v_block_sums(const bf16_t* __restrict__ V, float* __restrict__ blocksum)
{
  const int bh = blockIdx.x;
  const int b = bh >> 3, hh = bh & 7;
  const int d = threadIdx.x;
  for (int blk = 0; blk < 8; ++blk) {
    float s = 0.f;
    for (int j = 0; j < 64; ++j)
      s += bf2f(V[(size_t)(b * HIST_LEN + blk * 64 + j) * D_MODEL + hh * 64 + d]);
    blocksum[((size_t)bh * 8 + blk) * 64 + d] = s;
  }
}

// ---------------------------------------------------------------------------
// Fused residual + LayerNorm over last dim (512), bf16 in/out, f32 stats.
// grid = rows, block = 128; each thread owns 4 contiguous columns.
// ---------------------------------------------------------------------------
__global__ void __launch_bounds__(128)
ln_residual(const bf16_t* __restrict__ X, const bf16_t* __restrict__ R,
            const float* __restrict__ g, const float* __restrict__ be,
            bf16_t* __restrict__ Out)
{
  __shared__ float red[128];
  const int row = blockIdx.x, tid = threadIdx.x;
  const int c0 = tid * 4;
  v4bf xv = *(const v4bf*)(X + (size_t)row * D_MODEL + c0);
  v4bf rv = *(const v4bf*)(R + (size_t)row * D_MODEL + c0);
  float v[4];
  float s = 0.f;
#pragma unroll
  for (int i = 0; i < 4; ++i) { v[i] = bf2f(xv[i]) + bf2f(rv[i]); s += v[i]; }
  red[tid] = s; __syncthreads();
  for (int off = 64; off; off >>= 1) { if (tid < off) red[tid] += red[tid + off]; __syncthreads(); }
  float mu = red[0] * (1.f / D_MODEL);
  __syncthreads();
  float s2 = 0.f;
#pragma unroll
  for (int i = 0; i < 4; ++i) { float d = v[i] - mu; s2 += d * d; }
  red[tid] = s2; __syncthreads();
  for (int off = 64; off; off >>= 1) { if (tid < off) red[tid] += red[tid + off]; __syncthreads(); }
  float rstd = rsqrtf(red[0] * (1.f / D_MODEL) + 1e-6f);
  v4bf ov;
#pragma unroll
  for (int i = 0; i < 4; ++i)
    ov[i] = f2bf((v[i] - mu) * rstd * g[c0 + i] + be[c0 + i]);
  *(v4bf*)(Out + (size_t)row * D_MODEL + c0) = ov;
}

// ---------------------------------------------------------------------------
// Input projection: out[r,:] = x[r,0:8] @ W[8,512]  (f32 in, bf16 out)
// ---------------------------------------------------------------------------
__global__ void __launch_bounds__(256)
proj_in(const float* __restrict__ x, const float* __restrict__ w,
        bf16_t* __restrict__ out, int total)
{
  int idx = blockIdx.x * 256 + threadIdx.x;
  if (idx >= total) return;
  int r = idx >> 9, c = idx & 511;
  float s = 0.f;
#pragma unroll
  for (int f = 0; f < 8; ++f) s += x[r * 8 + f] * w[f * D_MODEL + c];
  out[idx] = f2bf(s);
}

// ---------------------------------------------------------------------------
// Final dense 512 -> 1 (bf16 in, f32 out). grid = rows, block = 128.
// ---------------------------------------------------------------------------
__global__ void __launch_bounds__(128)
final_dense(const bf16_t* __restrict__ X, const float* __restrict__ w,
            const float* __restrict__ b, float* __restrict__ out)
{
  __shared__ float red[128];
  const int row = blockIdx.x, tid = threadIdx.x;
  float s = 0.f;
#pragma unroll
  for (int i = 0; i < 4; ++i) {
    int d = tid + i * 128;
    s += bf2f(X[(size_t)row * D_MODEL + d]) * w[d];
  }
  red[tid] = s; __syncthreads();
  for (int off = 64; off; off >>= 1) { if (tid < off) red[tid] += red[tid + off]; __syncthreads(); }
  if (tid == 0) out[row] = red[0] + b[0];
}

// ---------------------------------------------------------------------------
// f32 -> bf16 conversion (weights), 4 elements per thread.
// ---------------------------------------------------------------------------
__global__ void __launch_bounds__(256)
cvt_f32_bf16(const float* __restrict__ in, bf16_t* __restrict__ out, int n4)
{
  int i = blockIdx.x * 256 + threadIdx.x;
  if (i >= n4) return;
  float4 f = ((const float4*)in)[i];
  v4bf o;
  o[0] = f2bf(f.x); o[1] = f2bf(f.y); o[2] = f2bf(f.z); o[3] = f2bf(f.w);
  ((v4bf*)out)[i] = o;
}

// ---------------------------------------------------------------------------
// Host orchestration
// ---------------------------------------------------------------------------
static inline void launch_gemm(hipStream_t st, const bf16_t* A, int lda, int seqMask,
                               int shift, const bf16_t* B, int ldb, const float* bias,
                               const bf16_t* Cin, bf16_t* C, int ldc,
                               int M, int N, int K, int act) {
  dim3 g(N / 64, M / 64);
  gemm_bf16<<<g, 128, 0, st>>>(A, lda, seqMask, shift, B, ldb, bias, Cin, C, ldc, M, N, K, act);
}

// causal conv1d (K=3) as three shifted, accumulating GEMMs
static inline void launch_conv3(hipStream_t st, const bf16_t* X, int seqMask,
                                const bf16_t* W, const float* bias, bf16_t* Y, int M) {
  launch_gemm(st, X, D_MODEL, seqMask, -2, W + 0 * D_MODEL * D_MODEL, D_MODEL, bias, nullptr, Y, D_MODEL, M, D_MODEL, D_MODEL, 0);
  launch_gemm(st, X, D_MODEL, seqMask, -1, W + 1 * D_MODEL * D_MODEL, D_MODEL, nullptr, Y, Y, D_MODEL, M, D_MODEL, D_MODEL, 0);
  launch_gemm(st, X, D_MODEL, seqMask,  0, W + 2 * D_MODEL * D_MODEL, D_MODEL, nullptr, Y, Y, D_MODEL, M, D_MODEL, D_MODEL, 0);
}

struct MHAb { const bf16_t *wq,*wk,*wv,*ow; const float *wqb,*wkb,*wvb,*ob; };

extern "C" void kernel_launch(void* const* d_in, const int* in_sizes, int n_in,
                              void* d_out, int out_size, void* d_ws, size_t ws_size,
                              hipStream_t stream) {
  (void)in_sizes; (void)n_in; (void)out_size; (void)ws_size;

  int ii = 0;
  auto nx = [&]() { return (const float*)d_in[ii++]; };

  // ---- workspace bump allocator (256B aligned) ----
  char* wsb = (char*)d_ws;
  auto alloc = [&](size_t bytes) {
    char* p = wsb; wsb += (bytes + 255) & ~(size_t)255; return (void*)p;
  };
  auto cvtW = [&](const float* src, size_t n) {
    bf16_t* dst = (bf16_t*)alloc(n * sizeof(bf16_t));
    int n4 = (int)(n / 4);
    cvt_f32_bf16<<<(n4 + 255) / 256, 256, 0, stream>>>(src, dst, n4);
    return (const bf16_t*)dst;
  };

  // setup_inputs() insertion order: x_enc, x_dec, params{enc_in, dec_in,
  // enc_layers, dec_layers, final}; each dict in insertion order.
  const float* x_enc  = nx();
  const float* x_dec  = nx();
  const float* enc_in = nx();
  const float* dec_in = nx();

  auto rdMHA = [&](int taps) {
    MHAb m;
    const float* w;
    w = nx(); m.wq = cvtW(w, (size_t)taps * D_MODEL * D_MODEL); m.wqb = nx();
    w = nx(); m.wk = cvtW(w, (size_t)D_MODEL * D_MODEL);        m.wkb = nx();
    w = nx(); m.wv = cvtW(w, (size_t)D_MODEL * D_MODEL);        m.wvb = nx();
    w = nx(); m.ow = cvtW(w, (size_t)D_MODEL * D_MODEL);        m.ob  = nx();
    return m;
  };
  auto rdFFN = [&](const bf16_t** f1w, const float** f1b,
                   const bf16_t** f2w, const float** f2b) {
    const float* w;
    w = nx(); *f1w = cvtW(w, (size_t)D_MODEL * DFF); *f1b = nx();
    w = nx(); *f2w = cvtW(w, (size_t)DFF * D_MODEL); *f2b = nx();
  };

  struct EncL { MHAb mha; const bf16_t *f1w,*f2w; const float *f1b,*f2b,*ln1g,*ln1b,*ln2g,*ln2b; } enc[2];
  for (int l = 0; l < 2; ++l) {
    enc[l].mha = rdMHA(3);
    rdFFN(&enc[l].f1w, &enc[l].f1b, &enc[l].f2w, &enc[l].f2b);
    enc[l].ln1g = nx(); enc[l].ln1b = nx();
    enc[l].ln2g = nx(); enc[l].ln2b = nx();
  }
  struct DecL { MHAb m1, m2; const bf16_t *f1w,*f2w;
                const float *f1b,*f2b,*ln1g,*ln1b,*ln2g,*ln2b,*ln3g,*ln3b; } dec[2];
  for (int l = 0; l < 2; ++l) {
    dec[l].m1 = rdMHA(3);
    dec[l].m2 = rdMHA(3);
    rdFFN(&dec[l].f1w, &dec[l].f1b, &dec[l].f2w, &dec[l].f2b);
    dec[l].ln1g = nx(); dec[l].ln1b = nx();
    dec[l].ln2g = nx(); dec[l].ln2b = nx();
    dec[l].ln3g = nx(); dec[l].ln3b = nx();
  }
  const float* fin_w = nx();
  const float* fin_b = nx();

  // ---- activation buffers (bf16) ----
  const size_t NE = (size_t)BATCH * HIST_LEN * D_MODEL;   // 8,388,608
  const size_t NM = (size_t)BATCH * HIST_LEN * DFF;       // 33,554,432
  const size_t ND = (size_t)BATCH * F_LEN * D_MODEL;      // 2,097,152
  bf16_t* E   = (bf16_t*)alloc(NE * 2);   // encoder activations (persist)
  bf16_t* Tq  = (bf16_t*)alloc(NE * 2);
  bf16_t* Tk  = (bf16_t*)alloc(NE * 2);
  bf16_t* Tv  = (bf16_t*)alloc(NE * 2);
  bf16_t* Ta  = (bf16_t*)alloc(NE * 2);
  bf16_t* Tm  = (bf16_t*)alloc(NM * 2);   // ffn intermediate
  bf16_t* Dd  = (bf16_t*)alloc(ND * 2);
  bf16_t* Do1 = (bf16_t*)alloc(ND * 2);
  bf16_t* Do2 = (bf16_t*)alloc(ND * 2);
  float*  Vsm = (float*)alloc((size_t)BATCH * NUM_HEADS * 8 * 64 * 4);

  const int Menc = BATCH * HIST_LEN;   // 16384
  const int Mdec = BATCH * F_LEN;      // 4096
  const int encMask = HIST_LEN - 1;
  const int decMask = F_LEN - 1;
  const dim3 attnEncGrid(BATCH * NUM_HEADS, 8);
  const dim3 attnDecGrid(BATCH * NUM_HEADS, F_LEN / 64);

  // ===================== Encoder =====================
  proj_in<<<(Menc * D_MODEL + 255) / 256, 256, 0, stream>>>(x_enc, enc_in, E, Menc * D_MODEL);

  for (int l = 0; l < 2; ++l) {
    const EncL& L = enc[l];
    launch_conv3(stream, E, encMask, L.mha.wq, L.mha.wqb, Tq, Menc);
    launch_gemm(stream, E, D_MODEL, encMask, 0, L.mha.wk, D_MODEL, L.mha.wkb, nullptr, Tk, D_MODEL, Menc, D_MODEL, D_MODEL, 0);
    launch_gemm(stream, E, D_MODEL, encMask, 0, L.mha.wv, D_MODEL, L.mha.wvb, nullptr, Tv, D_MODEL, Menc, D_MODEL, D_MODEL, 0);
    v_block_sums<<<BATCH * NUM_HEADS, 64, 0, stream>>>(Tv, Vsm);
    attn_sparse_enc<<<attnEncGrid, 128, 0, stream>>>(Tq, Tk, Tv, Vsm, Ta);
    launch_gemm(stream, Ta, D_MODEL, encMask, 0, L.mha.ow, D_MODEL, L.mha.ob, nullptr, Tq, D_MODEL, Menc, D_MODEL, D_MODEL, 0);
    ln_residual<<<Menc, 128, 0, stream>>>(E, Tq, L.ln1g, L.ln1b, E);
    launch_gemm(stream, E, D_MODEL, encMask, 0, L.f1w, DFF, L.f1b, nullptr, Tm, DFF, Menc, DFF, D_MODEL, 1 /*selu*/);
    launch_gemm(stream, Tm, DFF, encMask, 0, L.f2w, D_MODEL, L.f2b, nullptr, Ta, D_MODEL, Menc, D_MODEL, DFF, 0);
    ln_residual<<<Menc, 128, 0, stream>>>(E, Ta, L.ln2g, L.ln2b, E);
  }

  // ===================== Decoder =====================
  proj_in<<<(Mdec * D_MODEL + 255) / 256, 256, 0, stream>>>(x_dec, dec_in, Dd, Mdec * D_MODEL);

  for (int l = 0; l < 2; ++l) {
    const DecL& L = dec[l];
    // causal self-attention
    launch_conv3(stream, Dd, decMask, L.m1.wq, L.m1.wqb, Tq, Mdec);
    launch_gemm(stream, Dd, D_MODEL, decMask, 0, L.m1.wk, D_MODEL, L.m1.wkb, nullptr, Tk, D_MODEL, Mdec, D_MODEL, D_MODEL, 0);
    launch_gemm(stream, Dd, D_MODEL, decMask, 0, L.m1.wv, D_MODEL, L.m1.wvb, nullptr, Tv, D_MODEL, Mdec, D_MODEL, D_MODEL, 0);
    attn_dense<<<attnDecGrid, 128, 0, stream>>>(Tq, Tk, Tv, Ta, F_LEN, F_LEN, 1);
    launch_gemm(stream, Ta, D_MODEL, decMask, 0, L.m1.ow, D_MODEL, L.m1.ob, nullptr, Tq, D_MODEL, Mdec, D_MODEL, D_MODEL, 0);
    ln_residual<<<Mdec, 128, 0, stream>>>(Dd, Tq, L.ln1g, L.ln1b, Do1);
    // cross-attention (queries from Do1, keys/values from encoder E)
    launch_conv3(stream, Do1, decMask, L.m2.wq, L.m2.wqb, Tq, Mdec);
    launch_gemm(stream, E, D_MODEL, encMask, 0, L.m2.wk, D_MODEL, L.m2.wkb, nullptr, Tk, D_MODEL, Menc, D_MODEL, D_MODEL, 0);
    launch_gemm(stream, E, D_MODEL, encMask, 0, L.m2.wv, D_MODEL, L.m2.wvb, nullptr, Tv, D_MODEL, Menc, D_MODEL, D_MODEL, 0);
    attn_dense<<<attnDecGrid, 128, 0, stream>>>(Tq, Tk, Tv, Ta, F_LEN, HIST_LEN, 0);
    launch_gemm(stream, Ta, D_MODEL, decMask, 0, L.m2.ow, D_MODEL, L.m2.ob, nullptr, Tq, D_MODEL, Mdec, D_MODEL, D_MODEL, 0);
    ln_residual<<<Mdec, 128, 0, stream>>>(Do1, Tq, L.ln2g, L.ln2b, Do2);
    // FFN
    launch_gemm(stream, Do2, D_MODEL, decMask, 0, L.f1w, DFF, L.f1b, nullptr, Tm, DFF, Mdec, DFF, D_MODEL, 1 /*selu*/);
    launch_gemm(stream, Tm, DFF, decMask, 0, L.f2w, D_MODEL, L.f2b, nullptr, Ta, D_MODEL, Mdec, D_MODEL, DFF, 0);
    ln_residual<<<Mdec, 128, 0, stream>>>(Do2, Ta, L.ln3g, L.ln3b, Dd);
  }

  // ===================== Final dense (512 -> 1) =====================
  final_dense<<<Mdec, 128, 0, stream>>>(Dd, fin_w, fin_b, (float*)d_out);
}
